// Block_19224273617071
// MI455X (gfx1250) — matrix operand order repaired and checked by reading, not compiled, not run on hardware
//
#include <hip/hip_runtime.h>

// ---------------------------------------------------------------------------
// MI455X (gfx1250, wave32) Mamba block + FFN.
// All GEMMs via v_wmma_f32_16x16x32_bf16 (bf16 inputs, f32 accumulate).
// ---------------------------------------------------------------------------

typedef __bf16 bf16_t;
typedef __attribute__((ext_vector_type(16))) __bf16 v16bf;
typedef __attribute__((ext_vector_type(8)))  float  v8f;

static constexpr int Bb  = 2;
static constexpr int Ll  = 2048;
static constexpr int DM  = 1024;
static constexpr int DI  = 2048;   // d_inner
static constexpr int DS  = 16;     // d_state
static constexpr int DC  = 4;      // d_conv
static constexpr int DTR = 64;     // dt_rank
static constexpr int DF  = 4096;   // d_ff
static constexpr int ROWS = Bb * Ll;          // 4096
static constexpr int XDBL = DTR + 2 * DS;     // 96

__device__ __forceinline__ bf16_t f32_to_bf16(float f) {
  unsigned int u = __float_as_uint(f);
  unsigned int r = u + 0x7fffu + ((u >> 16) & 1u);   // round-to-nearest-even
  unsigned short h = (unsigned short)(r >> 16);
  return __builtin_bit_cast(bf16_t, h);
}

// ------------------------------- converts ----------------------------------
__global__ void cvt_f32_bf16_kernel(const float* __restrict__ in,
                                    bf16_t* __restrict__ out, long n) {
  long i = (long)blockIdx.x * blockDim.x + threadIdx.x;
  if (i < n) out[i] = f32_to_bf16(in[i]);
}

// slice cols [0,64) out of the 96-wide x_dbl rows -> bf16 dt matrix
__global__ void dt_convert_kernel(const float* __restrict__ xdbl,
                                  bf16_t* __restrict__ dtb) {
  int i = blockIdx.x * blockDim.x + threadIdx.x;
  if (i >= ROWS * DTR) return;
  int c = i % DTR, r = i / DTR;
  dtb[i] = f32_to_bf16(xdbl[(size_t)r * XDBL + c]);
}

// ------------------------------- layernorm ---------------------------------
__global__ __launch_bounds__(256)
void layernorm_bf16_kernel(const float* __restrict__ x, const float* __restrict__ w,
                           const float* __restrict__ b, bf16_t* __restrict__ out, int D) {
  __shared__ float s1[256], s2[256];
  int row = blockIdx.x;
  const float* xr = x + (size_t)row * D;
  float a = 0.f, q = 0.f;
  for (int i = threadIdx.x; i < D; i += 256) { float v = xr[i]; a += v; q += v * v; }
  s1[threadIdx.x] = a; s2[threadIdx.x] = q;
  __syncthreads();
  for (int s = 128; s > 0; s >>= 1) {
    if ((int)threadIdx.x < s) { s1[threadIdx.x] += s1[threadIdx.x + s];
                                s2[threadIdx.x] += s2[threadIdx.x + s]; }
    __syncthreads();
  }
  float mean = s1[0] / (float)D;
  float var  = s2[0] / (float)D - mean * mean;
  float rstd = rsqrtf(var + 1e-5f);
  for (int i = threadIdx.x; i < D; i += 256)
    out[(size_t)row * D + i] = f32_to_bf16((xr[i] - mean) * rstd * w[i] + b[i]);
}

// --------------------------- depthwise conv + SiLU -------------------------
__global__ __launch_bounds__(256)
void conv_silu_kernel(const float* __restrict__ xz, const float* __restrict__ cw,
                      const float* __restrict__ cb, float* __restrict__ uf,
                      bf16_t* __restrict__ ub) {
  int idx = blockIdx.x * blockDim.x + threadIdx.x;     // over ROWS*DI
  if (idx >= ROWS * DI) return;
  int d = idx % DI;
  int r = idx / DI;            // r = b*L + t
  int t = r % Ll;
  float acc = cb[d];
#pragma unroll
  for (int j = 0; j < DC; ++j) {
    int tt = t - (DC - 1) + j;
    if (tt >= 0)
      acc += cw[d * DC + j] * xz[(size_t)(r + j - (DC - 1)) * (2 * DI) + d];
  }
  float s = acc / (1.f + __expf(-acc));                // SiLU
  uf[(size_t)r * DI + d] = s;
  ub[(size_t)r * DI + d] = f32_to_bf16(s);
}

// ------------------------------- SSM scan ----------------------------------
// One thread per (batch, channel). 16-state recurrence kept in VGPRs.
// Fuses the D skip-term and the SiLU(z) gate; emits bf16 for out_proj WMMA.
__global__ __launch_bounds__(256)
void ssm_scan_kernel(const float* __restrict__ uf, const float* __restrict__ dlt,
                     const float* __restrict__ xdbl, const float* __restrict__ xz,
                     const float* __restrict__ A_log, const float* __restrict__ Dp,
                     bf16_t* __restrict__ yb) {
  int tid = blockIdx.x * blockDim.x + threadIdx.x;
  if (tid >= Bb * DI) return;
  int d = tid % DI;
  int b = tid / DI;
  float A[DS], h[DS];
#pragma unroll
  for (int n = 0; n < DS; ++n) { A[n] = -__expf(A_log[d * DS + n]); h[n] = 0.f; }
  float Dd = Dp[d];
  for (int t = 0; t < Ll; ++t) {
    size_t r = (size_t)b * Ll + t;
    float du = uf[r * DI + d];
    float dl = dlt[r * DI + d];
    const float4* Bv = reinterpret_cast<const float4*>(xdbl + r * XDBL + DTR);
    const float4* Cv = reinterpret_cast<const float4*>(xdbl + r * XDBL + DTR + DS);
    float y = 0.f;
    float dbu = dl * du;
#pragma unroll
    for (int q = 0; q < 4; ++q) {
      float4 bq = Bv[q], cq = Cv[q];
      float bb[4] = {bq.x, bq.y, bq.z, bq.w};
      float cc[4] = {cq.x, cq.y, cq.z, cq.w};
#pragma unroll
      for (int j = 0; j < 4; ++j) {
        int n = q * 4 + j;
        float dA = __expf(dl * A[n]);
        h[n] = h[n] * dA + dbu * bb[j];
        y += h[n] * cc[j];
      }
    }
    y += du * Dd;
    float z = xz[r * (2 * DI) + DI + d];
    y *= z / (1.f + __expf(-z));                       // * SiLU(z)
    yb[r * DI + d] = f32_to_bf16(y);
  }
}

// ------------------------------- WMMA GEMM ---------------------------------
// C(M,N) = A(M,K) @ W(N,K)^T ; A,W bf16 row-major, f32 accumulate.
// 8 waves/block, each wave does a 32x32 macro-tile (2x2 WMMA tiles, fragment
// reuse), block tile = 128x64. K always a multiple of 32; M a multiple of 128.
// GUARDN=true only for ragged N (x_proj, N=96) -> keeps the hot GEMMs' inner
// loop free of exec-mask saves and zero-fills.
// EPI: 0=none 1=bias+softplus 2=+residual 3=bias+relu 4=bias+accumulate(out+=)
struct ABFrag { union { v16bf v; uint4 q[2]; }; };

template <int EPI, int OUTB, bool GUARDN>
__global__ __launch_bounds__(256)
void wmma_gemm_kernel(const bf16_t* __restrict__ A, const bf16_t* __restrict__ W,
                      float* __restrict__ Cf, bf16_t* __restrict__ Cb,
                      const float* __restrict__ bias, const float* __restrict__ res,
                      int M, int N, int K) {
  const int lane = threadIdx.x & 31;
  const int wave = threadIdx.x >> 5;             // 0..7
  const int wm = wave >> 1, wn = wave & 1;       // 4x2 wave grid
  const int row0 = blockIdx.x * 128 + wm * 32;
  const int col0 = blockIdx.y * 64 + wn * 32;
  // ISA fragment layouts (CDNA5 ISA):
  //   A 16x32 bf16: lane<16 -> row M=lane, K pairs (0..7,16..23); lane>=16 -> K+8
  //   B 32x16 bf16 from row-major W(N,K): lane = column n, same K order as A
  const int khalf = (lane >> 4) * 8;             // A K offset within each 16-chunk
  const int ksel  = (lane >> 4) * 8;             // B K offset within each 16-chunk
  const int mA  = row0 + (lane & 15);
  const int nB0 = col0 + (lane & 15);
  const int nB1 = nB0 + 16;

  v8f acc00 = {}, acc01 = {}, acc10 = {}, acc11 = {};
  const uint4 zero4 = make_uint4(0u, 0u, 0u, 0u);

  for (int k0 = 0; k0 < K; k0 += 32) {
    ABFrag a0, a1, b0, b1;
    { const bf16_t* p = A + (size_t)mA * K + k0 + khalf;
      a0.q[0] = *(const uint4*)p; a0.q[1] = *(const uint4*)(p + 16); }
    { const bf16_t* p = A + (size_t)(mA + 16) * K + k0 + khalf;
      a1.q[0] = *(const uint4*)p; a1.q[1] = *(const uint4*)(p + 16); }
    if (!GUARDN || nB0 < N) {
      const bf16_t* p = W + (size_t)nB0 * K + k0 + ksel;
      b0.q[0] = *(const uint4*)p; b0.q[1] = *(const uint4*)(p + 16);
    } else { b0.q[0] = zero4; b0.q[1] = zero4; }
    if (!GUARDN || nB1 < N) {
      const bf16_t* p = W + (size_t)nB1 * K + k0 + ksel;
      b1.q[0] = *(const uint4*)p; b1.q[1] = *(const uint4*)(p + 16);
    } else { b1.q[0] = zero4; b1.q[1] = zero4; }

    acc00 = __builtin_amdgcn_wmma_f32_16x16x32_bf16(false, a0.v, false, b0.v,
                                                    (short)0, acc00, false, false);
    acc01 = __builtin_amdgcn_wmma_f32_16x16x32_bf16(false, a0.v, false, b1.v,
                                                    (short)0, acc01, false, false);
    acc10 = __builtin_amdgcn_wmma_f32_16x16x32_bf16(false, a1.v, false, b0.v,
                                                    (short)0, acc10, false, false);
    acc11 = __builtin_amdgcn_wmma_f32_16x16x32_bf16(false, a1.v, false, b1.v,
                                                    (short)0, acc11, false, false);
  }

  // D layout: lane = (m>=8 ? 16:0)+n ; vgpr v -> m = v + 8*(lane>=16)
  const int nc = lane & 15;
  const int mo = (lane >> 4) * 8;
  auto epi = [&](const v8f& acc, int mbase, int nbase) {
    int n = nbase + nc;
    if (GUARDN && n >= N) return;
#pragma unroll
    for (int v = 0; v < 8; ++v) {
      int m = mbase + mo + v;
      float val = acc[v];
      size_t o = (size_t)m * N + n;
      if constexpr (EPI == 1) {            // bias + softplus
        val += bias[n];
        val = (val > 20.f) ? val : __logf(1.f + __expf(val));
      } else if constexpr (EPI == 2) {     // + residual
        val += res[o];
      } else if constexpr (EPI == 3) {     // bias + relu
        val = fmaxf(val + bias[n], 0.f);
      } else if constexpr (EPI == 4) {     // bias + accumulate into existing out
        val += bias[n] + Cf[o];
      }
      if constexpr (OUTB) Cb[o] = f32_to_bf16(val);
      else                Cf[o] = val;
    }
  };
  epi(acc00, row0,      col0);
  epi(acc01, row0,      col0 + 16);
  epi(acc10, row0 + 16, col0);
  epi(acc11, row0 + 16, col0 + 16);
}

// ------------------------------- launcher ----------------------------------
extern "C" void kernel_launch(void* const* d_in, const int* in_sizes, int n_in,
                              void* d_out, int out_size, void* d_ws, size_t ws_size,
                              hipStream_t stream) {
  (void)in_sizes; (void)n_in; (void)out_size; (void)ws_size;
  const float* x         = (const float*)d_in[0];
  const float* ln1_w     = (const float*)d_in[1];
  const float* ln1_b     = (const float*)d_in[2];
  const float* ln2_w     = (const float*)d_in[3];
  const float* ln2_b     = (const float*)d_in[4];
  const float* in_proj_w = (const float*)d_in[5];
  const float* conv_w    = (const float*)d_in[6];
  const float* conv_b    = (const float*)d_in[7];
  const float* x_proj_w  = (const float*)d_in[8];
  const float* dt_proj_w = (const float*)d_in[9];
  const float* dt_proj_b = (const float*)d_in[10];
  const float* A_log     = (const float*)d_in[11];
  const float* Dp        = (const float*)d_in[12];
  const float* out_proj_w= (const float*)d_in[13];
  const float* ffn_w1    = (const float*)d_in[14];
  const float* ffn_b1    = (const float*)d_in[15];
  const float* ffn_w2    = (const float*)d_in[16];
  const float* ffn_b2    = (const float*)d_in[17];
  float* out = (float*)d_out;

  // ---- workspace carve-up (256B aligned) ----
  char* p = (char*)d_ws;
  auto alloc = [&](size_t bytes) {
    char* r = p; p += (bytes + 255) & ~(size_t)255; return r; };
  bf16_t* Win  = (bf16_t*)alloc((size_t)(2*DI)*DM * 2);   // 4096x1024
  bf16_t* Wxp  = (bf16_t*)alloc((size_t)XDBL*DI * 2);     // 96x2048
  bf16_t* Wdt  = (bf16_t*)alloc((size_t)DI*DTR * 2);      // 2048x64
  bf16_t* Wout = (bf16_t*)alloc((size_t)DM*DI * 2);       // 1024x2048
  bf16_t* Wf1  = (bf16_t*)alloc((size_t)DF*DM * 2);       // 4096x1024
  bf16_t* Wf2  = (bf16_t*)alloc((size_t)DM*DF * 2);       // 1024x4096
  bf16_t* hbf  = (bf16_t*)alloc((size_t)ROWS*DM * 2);     // ln1 out, later ln2 out
  float*  xz   = (float*) alloc((size_t)ROWS*(2*DI) * 4); // in_proj out (u|z)
  float*  uf   = (float*) alloc((size_t)ROWS*DI * 4);     // conv+silu f32
  bf16_t* ubf  = (bf16_t*)alloc((size_t)ROWS*DI * 2);     // conv+silu bf16
  float*  xdbl = (float*) alloc((size_t)ROWS*XDBL * 4);   // dt|B|C
  bf16_t* dtb  = (bf16_t*)alloc((size_t)ROWS*DTR * 2);
  float*  dlt  = (float*) alloc((size_t)ROWS*DI * 4);     // softplus(delta)
  bf16_t* ybf  = (bf16_t*)alloc((size_t)ROWS*DI * 2);     // gated scan out
  bf16_t* tbf  = (bf16_t*)xz;  // FFN hidden (32MB) aliases dead xz (64MB)

  auto cvt = [&](const float* in, bf16_t* o, long n) {
    cvt_f32_bf16_kernel<<<(unsigned)((n + 255) / 256), 256, 0, stream>>>(in, o, n);
  };
  // weights -> bf16 (stay hot in the 192MB L2 across the whole block)
  cvt(in_proj_w,  Win,  (long)(2*DI)*DM);
  cvt(x_proj_w,   Wxp,  (long)XDBL*DI);
  cvt(dt_proj_w,  Wdt,  (long)DI*DTR);
  cvt(out_proj_w, Wout, (long)DM*DI);
  cvt(ffn_w1,     Wf1,  (long)DF*DM);
  cvt(ffn_w2,     Wf2,  (long)DM*DF);

  // LN1 -> bf16
  layernorm_bf16_kernel<<<ROWS, 256, 0, stream>>>(x, ln1_w, ln1_b, hbf, DM);

  // xz = h1 @ in_proj_w^T   (4096 x 4096 x 1024)
  wmma_gemm_kernel<0,0,false><<<dim3(ROWS/128, (2*DI)/64), 256, 0, stream>>>(
      hbf, Win, xz, nullptr, nullptr, nullptr, ROWS, 2*DI, DM);

  // depthwise causal conv + SiLU
  conv_silu_kernel<<<(ROWS*DI)/256, 256, 0, stream>>>(xz, conv_w, conv_b, uf, ubf);

  // x_dbl = u @ x_proj_w^T  (4096 x 96 x 2048) -- ragged N, guarded
  wmma_gemm_kernel<0,0,true><<<dim3(ROWS/128, (XDBL+63)/64), 256, 0, stream>>>(
      ubf, Wxp, xdbl, nullptr, nullptr, nullptr, ROWS, XDBL, DI);

  // dt slice -> bf16
  dt_convert_kernel<<<(ROWS*DTR)/256, 256, 0, stream>>>(xdbl, dtb);

  // delta = softplus(dt @ dt_proj_w^T + b)  (4096 x 2048 x 64)
  wmma_gemm_kernel<1,0,false><<<dim3(ROWS/128, DI/64), 256, 0, stream>>>(
      dtb, Wdt, dlt, nullptr, dt_proj_b, nullptr, ROWS, DI, DTR);

  // selective scan (fused D-term + SiLU(z) gate), bf16 out
  ssm_scan_kernel<<<(Bb*DI)/256, 256, 0, stream>>>(uf, dlt, xdbl, xz, A_log, Dp, ybf);

  // x2 = x + y @ out_proj_w^T  (4096 x 1024 x 2048), residual fused
  wmma_gemm_kernel<2,0,false><<<dim3(ROWS/128, DM/64), 256, 0, stream>>>(
      ybf, Wout, out, nullptr, nullptr, x, ROWS, DM, DI);

  // LN2 -> bf16 (reuses hbf)
  layernorm_bf16_kernel<<<ROWS, 256, 0, stream>>>(out, ln2_w, ln2_b, hbf, DM);

  // t = relu(h2 @ w1^T + b1)  (4096 x 4096 x 1024), bf16 out
  wmma_gemm_kernel<3,1,false><<<dim3(ROWS/128, DF/64), 256, 0, stream>>>(
      hbf, Wf1, nullptr, tbf, ffn_b1, nullptr, ROWS, DF, DM);

  // out += t @ w2^T + b2  (4096 x 1024 x 4096), accumulate into residual
  wmma_gemm_kernel<4,0,false><<<dim3(ROWS/128, DM/64), 256, 0, stream>>>(
      tbf, Wf2, out, nullptr, ffn_b2, nullptr, ROWS, DM, DF);
}
